// Light_FusionAttention_74380243632196
// MI455X (gfx1250) — matrix operand order repaired and checked by reading, not compiled, not run on hardware
//
#include <hip/hip_runtime.h>

typedef __attribute__((ext_vector_type(16))) _Float16 v16h;
typedef __attribute__((ext_vector_type(8)))  _Float16 v8h;
typedef __attribute__((ext_vector_type(8)))  float    v8f;

#define B_   32
#define S_   2048
#define H_   8
#define DM_  512
#define BH_  (B_*H_)

// workspace layout (bytes)
#define WS_WT   0u              // 4 x 512x512 f16 transposed weights   (2 MB)
#define WS_QV   (2u<<20)        // 5 x B x H x 64 f32 folded queries    (320 KB)
#define WS_OH   (3u<<20)        // 4 x B x 512 f32 head outputs         (256 KB)
#define WS_SC   0x340000u       // 5 x B x H x S f32 raw scores         (10 MB)
#define WS_VTA  (16u<<20)       // B x 512 x S f16  v_a transposed      (64 MB)
#define WS_VTT  (80u<<20)       // B x 512 x S f16  v_t transposed      (64 MB)

__device__ __forceinline__ v8f wmma_f16(v16h a, v16h b, v8f c){
  return __builtin_amdgcn_wmma_f32_16x16x32_f16(false, a, false, b, (short)0, c, false, false);
}

// ---------------------------------------------------------------------------
// K0: transpose+convert WK_a, WK_t, WV_a, WV_t -> f16 WT[m][n][k]
//     m order: 0=WK_a 1=WK_t 2=WV_a 3=WV_t
// ---------------------------------------------------------------------------
__global__ __launch_bounds__(256) void prep_weights(
    const float* __restrict__ WKa, const float* __restrict__ WKt,
    const float* __restrict__ WVa, const float* __restrict__ WVt,
    _Float16* __restrict__ WT)
{
  size_t idx = (size_t)blockIdx.x*256 + threadIdx.x;   // 4 * 262144
  int m = (int)(idx >> 18);
  int r = (int)(idx & 262143);
  int n = r >> 9, k = r & 511;
  const float* W = (m==0) ? WKa : (m==1) ? WKt : (m==2) ? WVa : WVt;
  WT[idx] = (_Float16)W[(size_t)k*DM_ + n];
}

// ---------------------------------------------------------------------------
// K1: fold the key-fusion algebra onto the (length-1) queries.
//   Qv[0] = q_a            (· k_a)          Qv[2] = q_t          (· k_t)
//   Qv[1] = WF_t q_t       (· k_a)          Qv[3] = WF_a q_a     (· k_t)
//                                           Qv[4] = WF_a WF_t q_t(· k_t)
// layout f32 [5][B*H][64]
// ---------------------------------------------------------------------------
__global__ __launch_bounds__(256) void prep_queries(
    const float* __restrict__ a_last, const float* __restrict__ t_last,
    const float* __restrict__ WQa, const float* __restrict__ WQt,
    const float* __restrict__ WFa, const float* __restrict__ WFt,
    float* __restrict__ Qv)
{
  int b = blockIdx.x, tid = threadIdx.x;
  __shared__ float qa[512], qt[512], qt2s[512];
  const float* av = a_last + (size_t)b*DM_;
  const float* tv = t_last + (size_t)b*DM_;
  #pragma unroll
  for (int jj=0; jj<2; ++jj){
    int j = tid + jj*256;
    float sa = 0.f, st = 0.f;
    for (int k=0; k<512; ++k){
      sa += av[k]*WQa[(size_t)k*DM_ + j];
      st += tv[k]*WQt[(size_t)k*DM_ + j];
    }
    qa[j] = sa; qt[j] = st;
  }
  __syncthreads();
  float qa2v[2], qt2v[2];
  #pragma unroll
  for (int jj=0; jj<2; ++jj){
    int j = tid + jj*256, h = j >> 6, d = j & 63;
    float s1 = 0.f, s2 = 0.f;
    for (int e=0; e<64; ++e){
      s1 += WFa[d*64+e]*qa[h*64+e];   // (WF_a q_a)[d]
      s2 += WFt[d*64+e]*qt[h*64+e];   // (WF_t q_t)[d]
    }
    qa2v[jj] = s1; qt2v[jj] = s2; qt2s[j] = s2;
  }
  __syncthreads();
  #pragma unroll
  for (int jj=0; jj<2; ++jj){
    int j = tid + jj*256, h = j >> 6, d = j & 63;
    float s3 = 0.f;
    for (int e=0; e<64; ++e) s3 += WFa[d*64+e]*qt2s[h*64+e];  // (WF_a WF_t q_t)[d]
    size_t o = (size_t)(b*H_ + h)*64 + d;
    const size_t P = (size_t)BH_*64;
    Qv[0*P + o] = qa[j];
    Qv[1*P + o] = qt2v[jj];
    Qv[2*P + o] = qt[j];
    Qv[3*P + o] = qa2v[jj];
    Qv[4*P + o] = s3;
  }
}

// ---------------------------------------------------------------------------
// K2: fused projections + scores.  Grid (S/64, B), 128 threads, LDS-free.
// Each wave owns a 16-row tile; all 4 waves sweep the SAME weight matrix at
// the same time so B-fragments are shared through WGP$ (4x less L2 traffic).
// A-fragments (16 ksteps x 8 VGPRs) are converted f32->f16 straight into
// registers.  k_a/k_t never hit memory: per C-fragment each lane FMAs its
// column against the folded query vectors and a 4-step shfl_xor reduction
// emits the raw score rows.  v_a/v_t stored f16 transposed [b][dim][S].
// ---------------------------------------------------------------------------
__global__ __launch_bounds__(128) void fused_proj_score(
    const float* __restrict__ a, const float* __restrict__ t,
    const _Float16* __restrict__ WT, const float* __restrict__ Qv,
    _Float16* __restrict__ vTa, _Float16* __restrict__ vTt,
    float* __restrict__ scores)
{
  const int tid  = threadIdx.x;
  const int lane = tid & 31;
  const int wave = tid >> 5;
  const int b    = blockIdx.y;
  const int s0   = blockIdx.x*64 + wave*16;
  const int n    = lane & 15;
  const int hi   = lane >> 4;
  const int sel  = hi ? 8 : 0;
  const size_t QP = (size_t)BH_*64;

  #pragma unroll 1
  for (int src=0; src<2; ++src){
    // ---- load this wave's 16-row A tile as 16 WMMA fragments (128 VGPRs) ----
    const float* rowp = (src ? t : a) + ((size_t)b*S_ + s0 + n)*DM_;
    v16h af[16];
    #pragma unroll
    for (int ks=0; ks<16; ++ks){
      const int kb = ks*32;
      const float* p0 = rowp + kb + sel;        // K runs {kb..kb+7, kb+16..kb+23} (+8 for hi half)
      const float* p1 = rowp + kb + 16 + sel;
      float4 x0 = ((const float4*)p0)[0], x1 = ((const float4*)p0)[1];
      float4 y0 = ((const float4*)p1)[0], y1 = ((const float4*)p1)[1];
      v16h f;
      f[0]=(_Float16)x0.x;  f[1]=(_Float16)x0.y;  f[2]=(_Float16)x0.z;  f[3]=(_Float16)x0.w;
      f[4]=(_Float16)x1.x;  f[5]=(_Float16)x1.y;  f[6]=(_Float16)x1.z;  f[7]=(_Float16)x1.w;
      f[8]=(_Float16)y0.x;  f[9]=(_Float16)y0.y;  f[10]=(_Float16)y0.z; f[11]=(_Float16)y0.w;
      f[12]=(_Float16)y1.x; f[13]=(_Float16)y1.y; f[14]=(_Float16)y1.z; f[15]=(_Float16)y1.w;
      af[ks] = f;
    }

    // ---- pass 1: K projection, scores folded in, nothing stored but scores --
    {
      const _Float16* wt = WT + ((size_t)src << 18);     // WK_a / WK_t
      const int nty  = src ? 3 : 2;                      // #query vecs dotted
      const int tbase = src ? 2 : 0;                     // raw score array base
      float sc0[8], sc1[8], sc2[8];
      #pragma unroll
      for (int r=0;r<8;++r){ sc0[r]=0.f; sc1[r]=0.f; sc2[r]=0.f; }
      #pragma unroll 1
      for (int nt=0; nt<32; ++nt){
        v8f c = {0.f,0.f,0.f,0.f,0.f,0.f,0.f,0.f};
        #pragma unroll
        for (int ks=0; ks<16; ++ks){
          const int kb = ks*32;
          v16h bf;
          const _Float16* p = wt + (size_t)(nt*16 + n)*DM_ + kb + hi*8;
          ((v8h*)&bf)[0] = *(const v8h*)p;
          ((v8h*)&bf)[1] = *(const v8h*)(p + 16);
          c = wmma_f16(af[ks], bf, c);
        }
        // lane owns column e of head h; accumulate q·k partials
        const int h = nt >> 2;
        const int e = (nt & 3)*16 + n;
        const size_t qo = (size_t)(b*H_ + h)*64 + e;
        const float q0 = Qv[(size_t)(tbase+0)*QP + qo];
        const float q1 = Qv[(size_t)(tbase+1)*QP + qo];
        const float q2 = src ? Qv[(size_t)4*QP + qo] : 0.f;
        #pragma unroll
        for (int r=0;r<8;++r){
          sc0[r] += c[r]*q0;
          sc1[r] += c[r]*q1;
          sc2[r] += c[r]*q2;
        }
        if ((nt & 3) == 3){                 // head finished: reduce 16 lanes
          #pragma unroll
          for (int ty=0; ty<3; ++ty){
            if (ty >= nty) break;
            float* sc = (ty==0)? sc0 : (ty==1)? sc1 : sc2;
            #pragma unroll
            for (int r=0;r<8;++r){
              float x = sc[r];
              x += __shfl_xor(x, 1, 32);
              x += __shfl_xor(x, 2, 32);
              x += __shfl_xor(x, 4, 32);
              x += __shfl_xor(x, 8, 32);
              sc[r] = x;
            }
            if (n == 0){                    // lane0: rows 0-7, lane16: rows 8-15
              float* d = scores + ((size_t)(tbase+ty)*BH_ + (b*H_ + h))*S_ + s0 + hi*8;
              *(float4*)d       = make_float4(sc[0], sc[1], sc[2], sc[3]);
              *(float4*)(d + 4) = make_float4(sc[4], sc[5], sc[6], sc[7]);
            }
            #pragma unroll
            for (int r=0;r<8;++r) sc[r] = 0.f;
          }
        }
      }
    }

    // ---- pass 2: V projection -> global f16 [b][dim][S] ----
    {
      const _Float16* wt = WT + ((size_t)(2 + src) << 18);   // WV_a / WV_t
      _Float16* vT = src ? vTt : vTa;
      #pragma unroll 1
      for (int nt=0; nt<32; ++nt){
        v8f c = {0.f,0.f,0.f,0.f,0.f,0.f,0.f,0.f};
        #pragma unroll
        for (int ks=0; ks<16; ++ks){
          const int kb = ks*32;
          v16h bf;
          const _Float16* p = wt + (size_t)(nt*16 + n)*DM_ + kb + hi*8;
          ((v8h*)&bf)[0] = *(const v8h*)p;
          ((v8h*)&bf)[1] = *(const v8h*)(p + 16);
          c = wmma_f16(af[ks], bf, c);
        }
        v8h hv;
        #pragma unroll
        for (int r=0;r<8;++r) hv[r] = (_Float16)c[r];
        _Float16* vdst = vT + ((size_t)b*DM_ + nt*16 + n)*S_ + s0 + hi*8;
        *(v8h*)vdst = hv;                    // one contiguous 16B store
      }
    }
  }
}

// ---------------------------------------------------------------------------
// K3: per (b,h): combine raw scores -> 4 softmaxes -> 4 weighted sums over v.
// ---------------------------------------------------------------------------
__global__ __launch_bounds__(256) void softmax_av(
    const float* __restrict__ scores,
    const _Float16* __restrict__ vTa, const _Float16* __restrict__ vTt,
    float* __restrict__ oheads)
{
  const int bh = blockIdx.x, tid = threadIdx.x;
  const int b = bh >> 3, h = bh & 7;
  __shared__ float p[4][S_];        // [0]=c_a [1]=c_t [2]=c_a1 [3]=c_t1
  __shared__ float4 red[256];
  __shared__ float acc[4][2][64];

  const float* r0 = scores + (size_t)0*BH_*S_ + (size_t)bh*S_;
  const float* r1 = scores + (size_t)1*BH_*S_ + (size_t)bh*S_;
  const float* r2 = scores + (size_t)2*BH_*S_ + (size_t)bh*S_;
  const float* r3 = scores + (size_t)3*BH_*S_ + (size_t)bh*S_;
  const float* r4 = scores + (size_t)4*BH_*S_ + (size_t)bh*S_;

  const float scale = 0.125f;   // 1/sqrt(64)
  float4 lmax = make_float4(-1e30f, -1e30f, -1e30f, -1e30f);
  #pragma unroll
  for (int j=0; j<8; ++j){
    int s = j*256 + tid;
    float a0=r0[s], a1=r1[s], a2=r2[s], a3=r3[s], a4=r4[s];
    float ca  = scale*(a0 + a3);
    float ct  = scale*(a2 + a1 + a4);
    float ca1 = scale*a0;
    float ct1 = scale*a2;
    p[0][s]=ca; p[1][s]=ct; p[2][s]=ca1; p[3][s]=ct1;
    lmax.x=fmaxf(lmax.x,ca); lmax.y=fmaxf(lmax.y,ct);
    lmax.z=fmaxf(lmax.z,ca1); lmax.w=fmaxf(lmax.w,ct1);
  }
  red[tid]=lmax; __syncthreads();
  for (int off=128; off>0; off>>=1){
    if (tid<off){
      float4 o=red[tid+off], mm=red[tid];
      mm.x=fmaxf(mm.x,o.x); mm.y=fmaxf(mm.y,o.y); mm.z=fmaxf(mm.z,o.z); mm.w=fmaxf(mm.w,o.w);
      red[tid]=mm;
    }
    __syncthreads();
  }
  float4 m4 = red[0]; __syncthreads();

  float4 lsum = make_float4(0.f,0.f,0.f,0.f);
  #pragma unroll
  for (int j=0; j<8; ++j){
    int s = j*256 + tid;
    float e0=__expf(p[0][s]-m4.x); p[0][s]=e0; lsum.x+=e0;
    float e1=__expf(p[1][s]-m4.y); p[1][s]=e1; lsum.y+=e1;
    float e2=__expf(p[2][s]-m4.z); p[2][s]=e2; lsum.z+=e2;
    float e3=__expf(p[3][s]-m4.w); p[3][s]=e3; lsum.w+=e3;
  }
  red[tid]=lsum; __syncthreads();
  for (int off=128; off>0; off>>=1){
    if (tid<off){
      float4 o=red[tid+off], ss=red[tid];
      ss.x+=o.x; ss.y+=o.y; ss.z+=o.z; ss.w+=o.w;
      red[tid]=ss;
    }
    __syncthreads();
  }
  float4 s4 = red[0]; __syncthreads();

  // weighted sums: tid<128 audio (o_a,o_a1), tid>=128 text (o_t,o_t1)
  const int hf = tid >> 7, g = (tid >> 6) & 1, e = tid & 63;
  const _Float16* vsrc = (hf ? vTt : vTa)
                       + ((size_t)b*DM_ + h*64 + e)*S_ + g*1024;
  const float* p0 = p[hf];        // hf0: c_a   hf1: c_t
  const float* p1 = p[2 + hf];    // hf0: c_a1  hf1: c_t1
  float A0 = 0.f, A1 = 0.f;
  for (int s=0; s<1024; s+=8){
    v8h v8 = *(const v8h*)(vsrc + s);
    int sb = g*1024 + s;
    #pragma unroll
    for (int j=0; j<8; ++j){
      float vv = (float)v8[j];
      A0 += p0[sb+j]*vv;
      A1 += p1[sb+j]*vv;
    }
  }
  acc[hf][g][e]   = A0;      // v=0 (o_a) / v=1 (o_t)
  acc[2+hf][g][e] = A1;      // v=2 (o_a1)/ v=3 (o_t1)
  __syncthreads();

  const int v = tid >> 6, ee = tid & 63;
  float tot = acc[v][0][ee] + acc[v][1][ee];
  float inv = 1.f / ((v==0) ? s4.x : (v==1) ? s4.y : (v==2) ? s4.z : s4.w);
  oheads[((size_t)v*BH_ + bh)*64 + ee] = tot*inv;
}

// ---------------------------------------------------------------------------
// K4: output projections  out[v][b] = oh[v][b] @ WO_{a,t}   (tiny)
// ---------------------------------------------------------------------------
__global__ __launch_bounds__(256) void out_proj(
    const float* __restrict__ oheads,
    const float* __restrict__ WOa, const float* __restrict__ WOt,
    float* __restrict__ out)
{
  const int v = blockIdx.x >> 5;    // 0:o_a 1:o_t 2:o_a1 3:o_t1 (return order)
  const int b = blockIdx.x & 31;
  const int tid = threadIdx.x;
  __shared__ float oh[512];
  const float* src = oheads + ((size_t)v*B_ + b)*DM_;
  oh[tid] = src[tid]; oh[tid+256] = src[tid+256];
  __syncthreads();
  const float* WO = (v & 1) ? WOt : WOa;
  #pragma unroll
  for (int jj=0; jj<2; ++jj){
    int j = tid + jj*256;
    float acc2 = 0.f;
    #pragma unroll 8
    for (int i=0; i<512; ++i) acc2 += oh[i]*WO[(size_t)i*DM_ + j];
    out[((size_t)v*B_ + b)*DM_ + j] = acc2;
  }
}

// ---------------------------------------------------------------------------
extern "C" void kernel_launch(void* const* d_in, const int* in_sizes, int n_in,
                              void* d_out, int out_size, void* d_ws, size_t ws_size,
                              hipStream_t stream)
{
  (void)in_sizes; (void)n_in; (void)out_size; (void)ws_size;
  const float* a      = (const float*)d_in[0];
  const float* t      = (const float*)d_in[1];
  const float* a_last = (const float*)d_in[2];
  const float* t_last = (const float*)d_in[3];
  const float* WQa = (const float*)d_in[4];
  const float* WKa = (const float*)d_in[5];
  const float* WVa = (const float*)d_in[6];
  const float* WOa = (const float*)d_in[7];
  const float* WQt = (const float*)d_in[8];
  const float* WKt = (const float*)d_in[9];
  const float* WVt = (const float*)d_in[10];
  const float* WOt = (const float*)d_in[11];
  const float* WFa = (const float*)d_in[12];
  const float* WFt = (const float*)d_in[13];

  char* ws = (char*)d_ws;
  _Float16* WT  = (_Float16*)(ws + WS_WT);
  float*    Qv  = (float*)   (ws + WS_QV);
  float*    oh  = (float*)   (ws + WS_OH);
  float*    sc  = (float*)   (ws + WS_SC);
  _Float16* vTa = (_Float16*)(ws + WS_VTA);
  _Float16* vTt = (_Float16*)(ws + WS_VTT);
  float* out = (float*)d_out;

  prep_weights    <<<4096, 256, 0, stream>>>(WKa, WKt, WVa, WVt, WT);
  prep_queries    <<<B_,   256, 0, stream>>>(a_last, t_last, WQa, WQt, WFa, WFt, Qv);
  fused_proj_score<<<dim3(S_/64, B_), 128, 0, stream>>>(a, t, WT, Qv, vTa, vTt, sc);
  softmax_av      <<<BH_,  256, 0, stream>>>(sc, vTa, vTt, oh);
  out_proj        <<<4*B_, 256, 0, stream>>>(oh, WOa, WOt, out);
}
